// HomoSAGE_22548578304459
// MI455X (gfx1250) — matrix operand order repaired and checked by reading, not compiled, not run on hardware
//
#include <hip/hip_runtime.h>

// ---------------------------------------------------------------------------
// GraphSAGE 2-layer forward for MI455X (gfx1250, wave32, WMMA).
//   Layer: out = relu( mean_agg(x) @ Wl^T + b + x @ Wr^T )
// Strategy:
//   - degree + scatter-add aggregation with f32 global atomics (L2-resident)
//   - bf16 operand copies, fp32 accumulation via v_wmma_f32_16x16x32_bf16
//   - A-panels staged in LDS (ds_store/ds_load b128), B from L0/L2
// ---------------------------------------------------------------------------

typedef __attribute__((ext_vector_type(16))) __bf16 v16bf;
typedef __attribute__((ext_vector_type(8)))  __bf16 v8bf;
typedef __attribute__((ext_vector_type(8)))  float  v8f;

#define NN    100000
#define EE    640000
#define DIN   128
#define HH    256

// ---------------- utility kernels -----------------------------------------

__global__ __launch_bounds__(256) void zero_f32(float* __restrict__ p, int n4) {
  int i = blockIdx.x * 256 + threadIdx.x;
  if (i < n4) {
    float4 z = {0.f, 0.f, 0.f, 0.f};
    ((float4*)p)[i] = z;
  }
}

__global__ __launch_bounds__(256) void cvt_f32_bf16(const float* __restrict__ in,
                                                    __bf16* __restrict__ out, int n) {
  int i = blockIdx.x * 256 + threadIdx.x;
  if (i < n) out[i] = (__bf16)in[i];
}

__global__ __launch_bounds__(256) void degree_k(const int* __restrict__ dst,
                                                float* __restrict__ deg, int e) {
  int i = blockIdx.x * 256 + threadIdx.x;
  if (i < e)
    __hip_atomic_fetch_add(&deg[dst[i]], 1.0f, __ATOMIC_RELAXED,
                           __HIP_MEMORY_SCOPE_AGENT);
}

// scatter-add of f32 features: one thread = (edge, 4-feature chunk)
__global__ __launch_bounds__(256) void scatter_f32(const float* __restrict__ feat,
                                                   const int* __restrict__ src,
                                                   const int* __restrict__ dst,
                                                   float* __restrict__ agg,
                                                   int e, int d4, int d) {
  int i = blockIdx.x * 256 + threadIdx.x;
  int ed = i / d4, c = i - ed * d4;
  if (ed >= e) return;
  int s = src[ed], t = dst[ed];
  float4 v = *(const float4*)(feat + (size_t)s * d + c * 4);
  float* o = agg + (size_t)t * d + c * 4;
  __hip_atomic_fetch_add(o + 0, v.x, __ATOMIC_RELAXED, __HIP_MEMORY_SCOPE_AGENT);
  __hip_atomic_fetch_add(o + 1, v.y, __ATOMIC_RELAXED, __HIP_MEMORY_SCOPE_AGENT);
  __hip_atomic_fetch_add(o + 2, v.z, __ATOMIC_RELAXED, __HIP_MEMORY_SCOPE_AGENT);
  __hip_atomic_fetch_add(o + 3, v.w, __ATOMIC_RELAXED, __HIP_MEMORY_SCOPE_AGENT);
}

// scatter-add of bf16 features (layer-2 h1), accumulate f32
__global__ __launch_bounds__(256) void scatter_bf16(const __bf16* __restrict__ feat,
                                                    const int* __restrict__ src,
                                                    const int* __restrict__ dst,
                                                    float* __restrict__ agg,
                                                    int e, int d4, int d) {
  int i = blockIdx.x * 256 + threadIdx.x;
  int ed = i / d4, c = i - ed * d4;
  if (ed >= e) return;
  int s = src[ed], t = dst[ed];
  const __bf16* p = feat + (size_t)s * d + c * 4;
  float* o = agg + (size_t)t * d + c * 4;
  __hip_atomic_fetch_add(o + 0, (float)p[0], __ATOMIC_RELAXED, __HIP_MEMORY_SCOPE_AGENT);
  __hip_atomic_fetch_add(o + 1, (float)p[1], __ATOMIC_RELAXED, __HIP_MEMORY_SCOPE_AGENT);
  __hip_atomic_fetch_add(o + 2, (float)p[2], __ATOMIC_RELAXED, __HIP_MEMORY_SCOPE_AGENT);
  __hip_atomic_fetch_add(o + 3, (float)p[3], __ATOMIC_RELAXED, __HIP_MEMORY_SCOPE_AGENT);
}

// mean = agg / max(deg,1), rounded to bf16
__global__ __launch_bounds__(256) void mean_cvt(const float* __restrict__ agg,
                                                const float* __restrict__ deg,
                                                __bf16* __restrict__ meanb,
                                                int n_elems, int d) {
  int i = blockIdx.x * 256 + threadIdx.x;
  if (i >= n_elems) return;
  int node = i / d;
  float inv = 1.0f / fmaxf(deg[node], 1.0f);
  meanb[i] = (__bf16)(agg[i] * inv);
}

// ---------------- fused SAGE GEMM ------------------------------------------
// out[16 rows x Hout] per block; 8 wave32s, each wave owns 2 column tiles.
// out = relu( Am @ Wl^T + bias + Ax @ Wr^T ),  Am/Ax: [N x K] bf16 row-major,
// Wl/Wr: [Hout x K] bf16 row-major (so B[k][n] = W[n][k] is contiguous in k).
template <typename OutT>
__global__ __launch_bounds__(256) void sage_gemm(const __bf16* __restrict__ Am,
                                                 const __bf16* __restrict__ Ax,
                                                 const __bf16* __restrict__ Wl,
                                                 const __bf16* __restrict__ Wr,
                                                 const float* __restrict__ bias,
                                                 OutT* __restrict__ out,
                                                 int K, int Hout) {
  __shared__ __bf16 ldsA[2 * 16 * 256];  // two 16xK panels (K<=256) = 16 KB

  const int tid = threadIdx.x;
  const int rowBase = blockIdx.x * 16;
  const int k8 = K >> 3;

  // cooperative stage of both A panels into LDS (b128 stores)
  for (int idx = tid; idx < 16 * k8; idx += 256) {
    int r = idx / k8, c = idx - r * k8;
    *(v8bf*)&ldsA[r * K + c * 8] =
        *(const v8bf*)(Am + (size_t)(rowBase + r) * K + c * 8);
    *(v8bf*)&ldsA[16 * K + r * K + c * 8] =
        *(const v8bf*)(Ax + (size_t)(rowBase + r) * K + c * 8);
  }
  __syncthreads();

  const int lane = tid & 31;
  const int wave = tid >> 5;
  const int half = (lane < 16) ? 0 : 1;  // wave32 half select
  const int lr = lane & 15;
  const int tn0 = wave * 32;             // this wave's two 16-col tiles

  float bv0 = bias[tn0 + lr];
  float bv1 = bias[tn0 + 16 + lr];
  v8f acc0, acc1;
#pragma unroll
  for (int j = 0; j < 8; ++j) { acc0[j] = bv0; acc1[j] = bv1; }

#pragma unroll
  for (int pass = 0; pass < 2; ++pass) {
    const __bf16* A = &ldsA[pass * 16 * K];
    const __bf16* W = pass ? Wr : Wl;
    for (int kk = 0; kk < K; kk += 32) {
      // A fragment (16-bit 16x32 layout): lane holds row lr;
      // elems 0..15 at K = kk + half*16 + s (contiguous, matches B)
      const __bf16* ap = A + lr * K + kk + half * 16;
      v8bf lo = *(const v8bf*)ap;
      v8bf hi = *(const v8bf*)(ap + 8);
      v16bf a;
#pragma unroll
      for (int j = 0; j < 8; ++j) { a[j] = lo[j]; a[8 + j] = hi[j]; }

      // B fragments: lane = column n, 16 consecutive K values per half
      int koff = kk + half * 16;
      v16bf b0 = *(const v16bf*)(W + (size_t)(tn0 + lr) * K + koff);
      v16bf b1 = *(const v16bf*)(W + (size_t)(tn0 + 16 + lr) * K + koff);

      acc0 = __builtin_amdgcn_wmma_f32_16x16x32_bf16(false, a, false, b0,
                                                     (short)0, acc0, false, false);
      acc1 = __builtin_amdgcn_wmma_f32_16x16x32_bf16(false, a, false, b1,
                                                     (short)0, acc1, false, false);
    }
  }

  // C/D layout: VGPR j -> M = j + half*8, N = lr. Fused ReLU on store.
#pragma unroll
  for (int j = 0; j < 8; ++j) {
    size_t row = (size_t)(rowBase + j + half * 8);
    out[row * Hout + tn0 + lr]      = (OutT)fmaxf(acc0[j], 0.0f);
    out[row * Hout + tn0 + 16 + lr] = (OutT)fmaxf(acc1[j], 0.0f);
  }
}

// ---------------- host-side launch -----------------------------------------

extern "C" void kernel_launch(void* const* d_in, const int* in_sizes, int n_in,
                              void* d_out, int out_size, void* d_ws, size_t ws_size,
                              hipStream_t stream) {
  const float* x   = (const float*)d_in[0];
  const int*   ei  = (const int*)d_in[1];
  const float* W1l = (const float*)d_in[2];
  const float* b1  = (const float*)d_in[3];
  const float* W1r = (const float*)d_in[4];
  const float* W2l = (const float*)d_in[5];
  const float* b2  = (const float*)d_in[6];
  const float* W2r = (const float*)d_in[7];
  float* out = (float*)d_out;

  const int* src = ei;
  const int* dst = ei + EE;

  const size_t MB = 1024ull * 1024ull;
  char* ws = (char*)d_ws;
  float*  deg   = (float*)(ws);              // 0.4 MB
  float*  agg   = (float*)(ws + 1 * MB);     // 102.4 MB (N x 256 f32)
  __bf16* meanb = (__bf16*)(ws + 104 * MB);  // 51.2 MB
  __bf16* h1b   = (__bf16*)(ws + 156 * MB);  // 51.2 MB
  __bf16* x0b   = (__bf16*)(ws + 208 * MB);  // 25.6 MB
  __bf16* w1lb  = (__bf16*)(ws + 234 * MB);  // weights (bf16)
  __bf16* w1rb  = w1lb + HH * DIN;
  __bf16* w2lb  = w1rb + HH * DIN;
  __bf16* w2rb  = w2lb + HH * HH;

  auto blocks = [](long long n) { return (unsigned)((n + 255) / 256); };

  // ---- one-time per call: degree + bf16 operand copies
  zero_f32<<<blocks(NN / 4), 256, 0, stream>>>(deg, NN / 4);
  zero_f32<<<blocks((long long)NN * DIN / 4), 256, 0, stream>>>(agg, NN * DIN / 4);
  degree_k<<<blocks(EE), 256, 0, stream>>>(dst, deg, EE);
  cvt_f32_bf16<<<blocks((long long)NN * DIN), 256, 0, stream>>>(x, x0b, NN * DIN);
  cvt_f32_bf16<<<blocks(HH * DIN), 256, 0, stream>>>(W1l, w1lb, HH * DIN);
  cvt_f32_bf16<<<blocks(HH * DIN), 256, 0, stream>>>(W1r, w1rb, HH * DIN);
  cvt_f32_bf16<<<blocks(HH * HH), 256, 0, stream>>>(W2l, w2lb, HH * HH);
  cvt_f32_bf16<<<blocks(HH * HH), 256, 0, stream>>>(W2r, w2rb, HH * HH);

  // ---- layer 1: agg(x) -> mean -> GEMM -> h1 (bf16)
  scatter_f32<<<blocks((long long)EE * (DIN / 4)), 256, 0, stream>>>(
      x, src, dst, agg, EE, DIN / 4, DIN);
  mean_cvt<<<blocks((long long)NN * DIN), 256, 0, stream>>>(
      agg, deg, meanb, NN * DIN, DIN);
  sage_gemm<__bf16><<<NN / 16, 256, 0, stream>>>(
      meanb, x0b, w1lb, w1rb, b1, h1b, DIN, HH);

  // ---- layer 2: agg(h1) -> mean -> GEMM -> out (f32)
  zero_f32<<<blocks((long long)NN * HH / 4), 256, 0, stream>>>(agg, NN * HH / 4);
  scatter_bf16<<<blocks((long long)EE * (HH / 4)), 256, 0, stream>>>(
      h1b, src, dst, agg, EE, HH / 4, HH);
  mean_cvt<<<blocks((long long)NN * HH), 256, 0, stream>>>(
      agg, deg, meanb, NN * HH, HH);
  sage_gemm<float><<<NN / 16, 256, 0, stream>>>(
      meanb, h1b, w2lb, w2rb, b2, out, HH, HH);
}
